// VQLayer_76596446756889
// MI455X (gfx1250) — matrix operand order repaired and checked by reading, not compiled, not run on hardware
//
#include <hip/hip_runtime.h>
#include <hip/hip_bf16.h>
#include <cstdint>

typedef __attribute__((ext_vector_type(16))) _Float16 v16h;
typedef __attribute__((ext_vector_type(8)))  float    v8f;

#define NUM_EMB   1024
#define EMB_DIM   256
#define N_TOKENS  32768        // 32 * 32 * 32
#define HW        1024         // 32*32
#define Q_ELEMS   8388608      // 32*256*32*32

// ---------------- ws layout ----------------
#define WS_CBP    0                      // u32[(k/2)*1024 + n]   (512 KB)
#define WS_CNORM  (512*1024)             // f32[1024]
#define WS_COUNTS (WS_CNORM + 4096)      // f32[1024]
#define WS_IDX    (WS_COUNTS + 4096)     // int[32768]            (128 KB)
#define WS_BSUM   (WS_IDX + 131072)      // f32[1024]

// Async global->LDS staging (gfx1250 ASYNCcnt path), guarded so we fall back
// to register staging + ds_store if the builtins aren't available.
#if defined(__has_builtin)
# if __has_builtin(__builtin_amdgcn_global_load_async_to_lds_b32) && \
     __has_builtin(__builtin_amdgcn_s_wait_asynccnt)
#  define USE_ASYNC_LDS 1
# endif
#endif
#ifndef USE_ASYNC_LDS
# define USE_ASYNC_LDS 0
#endif

#if USE_ASYNC_LDS
// builtin signature (from hipcc diagnostic): arg0 is '__device__ int *'
// (addrspace(1), non-const, int); make the LDS side addrspace(3) int* to match.
typedef __attribute__((address_space(1))) int glob_i32_t;
typedef __attribute__((address_space(3))) int lds_i32_t;
#endif

// ---------------------------------------------------------------------------
// Kernel A: codebook -> f16 packed-transposed layout + ||c||^2 + zero counts
// grid = 1024 (one block per codebook entry), block = 128 (one k-pair each)
// ---------------------------------------------------------------------------
__global__ __launch_bounds__(128) void vq_prep_kernel(
    const float* __restrict__ cb, uint32_t* __restrict__ cbP,
    float* __restrict__ cnorm, float* __restrict__ counts)
{
  const int n = blockIdx.x;
  const int t = threadIdx.x;                 // k-pair index: k = 2t, 2t+1
  const float c0 = cb[n * EMB_DIM + 2 * t];
  const float c1 = cb[n * EMB_DIM + 2 * t + 1];
  union { uint32_t u; _Float16 h[2]; } p;
  p.h[0] = (_Float16)c0;
  p.h[1] = (_Float16)c1;
  cbP[(size_t)t * NUM_EMB + n] = p.u;        // B-fragment friendly layout

  __shared__ float red[128];
  red[t] = c0 * c0 + c1 * c1;
  __syncthreads();
  for (int st = 64; st > 0; st >>= 1) {
    if (t < st) red[t] += red[t + st];
    __syncthreads();
  }
  if (t == 0) {
    cnorm[n]  = red[0];
    counts[n] = 0.0f;                        // re-zero histogram every call
  }
}

// ---------------------------------------------------------------------------
// Kernel B: WMMA distance GEMM + argmin, B-tiles staged through LDS
// (double-buffered, async DMA when available) shared by the block's 8 waves.
// grid = 256 blocks x 256 threads; each wave owns 16 tokens, scans all 1024
// codebook entries with v_wmma_f32_16x16x32_f16.
// argmin target: ||c||^2 - 2 x.c   (||x||^2 is row-constant)
// ---------------------------------------------------------------------------
__global__ __launch_bounds__(256) void vq_argmin_kernel(
    const float*    __restrict__ inputs,   // NCHW f32
    const uint32_t* __restrict__ cbP,
    const float*    __restrict__ cnorm,
    int*   __restrict__ idxOut,
    float* __restrict__ idxOutF,
    float* __restrict__ counts)
{
  // [buf][n-in-tile][k2] ; row pad 132 dwords keeps b128 frags 16B aligned
  // and spreads banks (132 % 64 == 4).
  __shared__ __align__(16) uint32_t ldsB[2][16][132];

  const int tid   = threadIdx.x;
  const int lane  = tid & 31;
  const int wave  = tid >> 5;
  const int half_ = lane >> 4;               // 0: lanes 0-15, 1: lanes 16-31
  const int l16   = lane & 15;
  const int tokBase = blockIdx.x * 128 + wave * 16;   // 16 tokens / wave
  const int b  = tokBase >> 10;              // / 1024 (HW)
  const int hw = tokBase & 1023;

  // cooperative B-tile fetch coordinates: 16 consecutive n per 16-lane group
  const int fn = tid & 15;                   // n within tile
  const int fk = tid >> 4;                   // k2 base 0..15 (rows fk+16*i)

  // ---- A fragments: 16 tokens x 256 dims as 8 chunks of 16x32 (contiguous k = 16*half + s)
  const float* aBase = inputs + (size_t)b * EMB_DIM * HW + hw + l16;
  v16h aF[8];
#pragma unroll
  for (int kc = 0; kc < 8; ++kc) {
#pragma unroll
    for (int v = 0; v < 8; ++v) {
      const int kb = 16 * half_ + 2 * v;
      const int K  = kc * 32 + kb;
      aF[kc][2 * v]     = (_Float16)aBase[(size_t)K * HW];
      aF[kc][2 * v + 1] = (_Float16)aBase[(size_t)(K + 1) * HW];
    }
  }

#if USE_ASYNC_LDS
  // ---- prologue: async DMA tile 0 -> buf0 and tile 1 -> buf1
#pragma unroll
  for (int i = 0; i < 8; ++i)
    __builtin_amdgcn_global_load_async_to_lds_b32(
        (glob_i32_t*)(cbP + (size_t)(fk + 16 * i) * NUM_EMB + fn),
        (lds_i32_t*)&ldsB[0][fn][fk + 16 * i], 0, 0);
#pragma unroll
  for (int i = 0; i < 8; ++i)
    __builtin_amdgcn_global_load_async_to_lds_b32(
        (glob_i32_t*)(cbP + (size_t)(fk + 16 * i) * NUM_EMB + 16 + fn),
        (lds_i32_t*)&ldsB[1][fn][fk + 16 * i], 0, 0);
  __builtin_amdgcn_s_wait_asynccnt(8);       // tile 0 complete (in-order)
  __syncthreads();
#else
  // ---- prologue: tile 0 -> LDS buf0 (via regs), tile 1 -> staging regs
  uint32_t stage[8];
#pragma unroll
  for (int i = 0; i < 8; ++i)
    ldsB[0][fn][fk + 16 * i] = cbP[(size_t)(fk + 16 * i) * NUM_EMB + fn];
#pragma unroll
  for (int i = 0; i < 8; ++i)
    stage[i] = cbP[(size_t)(fk + 16 * i) * NUM_EMB + 16 + fn];
  __syncthreads();
#endif

  float best[8];
  int   bestN[8];
#pragma unroll
  for (int r = 0; r < 8; ++r) { best[r] = 3.0e38f; bestN[r] = 0x7fffffff; }

  for (int nt = 0; nt < 64; ++nt) {          // 64 tiles of 16 codebook entries
    const int N = nt * 16 + l16;
    const uint32_t* rowp = &ldsB[nt & 1][l16][8 * half_];
    v8f acc = {};
#pragma unroll
    for (int kc = 0; kc < 8; ++kc) {
      // lane's B fragment: 8 contiguous dwords -> two ds_load_b128
      union { uint4 q[2]; v16h h; } u;
      u.q[0] = *(const uint4*)(rowp + 16 * kc);
      u.q[1] = *(const uint4*)(rowp + 16 * kc + 4);
      acc = __builtin_amdgcn_wmma_f32_16x16x32_f16(
          /*neg_a=*/false, aF[kc], /*neg_b=*/false, u.h,
          /*c_mod=*/(short)0, acc, /*reuse_a=*/false, /*reuse_b=*/false);
    }
    const float cn = cnorm[N];
#pragma unroll
    for (int r = 0; r < 8; ++r) {            // D[r] = row M = r + 8*half
      const float d = cn - 2.0f * acc[r];
      if (d < best[r]) { best[r] = d; bestN[r] = N; }
    }

#if USE_ASYNC_LDS
    if (nt < 62) {
      __syncthreads();                       // all waves done reading buf[nt&1]
#pragma unroll
      for (int i = 0; i < 8; ++i)            // DMA tile nt+2 into buf[nt&1]
        __builtin_amdgcn_global_load_async_to_lds_b32(
            (glob_i32_t*)(cbP + (size_t)(fk + 16 * i) * NUM_EMB +
                          (nt + 2) * 16 + fn),
            (lds_i32_t*)&ldsB[nt & 1][fn][fk + 16 * i], 0, 0);
      __builtin_amdgcn_s_wait_asynccnt(8);   // tile nt+1 (older 8 ops) done
      __syncthreads();                       // visible block-wide
    } else if (nt == 62) {
      __syncthreads();
      __builtin_amdgcn_s_wait_asynccnt(0);   // tile 63 done
      __syncthreads();
    }
#else
    if (nt < 63) {
      __syncthreads();                       // everyone done reading buf^1
#pragma unroll
      for (int i = 0; i < 8; ++i)
        ldsB[(nt + 1) & 1][fn][fk + 16 * i] = stage[i];
      if (nt < 62) {
#pragma unroll
        for (int i = 0; i < 8; ++i)
          stage[i] = cbP[(size_t)(fk + 16 * i) * NUM_EMB + (nt + 2) * 16 + fn];
      }
      __syncthreads();                       // tile nt+1 ready
    }
#endif
  }

  // reduce argmin across the 16 lanes of each half-group (first-min tiebreak)
#pragma unroll
  for (int off = 8; off >= 1; off >>= 1) {
#pragma unroll
    for (int r = 0; r < 8; ++r) {
      const float od = __shfl_xor(best[r],  off, 32);
      const int   on = __shfl_xor(bestN[r], off, 32);
      if (od < best[r] || (od == best[r] && on < bestN[r])) {
        best[r] = od; bestN[r] = on;
      }
    }
  }

  if (l16 == 0) {                            // lanes 0 and 16 hold 8 rows each
#pragma unroll
    for (int r = 0; r < 8; ++r) {
      const int token = tokBase + r + 8 * half_;
      const int n = bestN[r];
      idxOut[token]  = n;
      idxOutF[token] = (float)n;
      atomicAdd(&counts[n], 1.0f);           // exact integer counts: order-safe
    }
  }
}

// ---------------------------------------------------------------------------
// Kernel C: gather quantized output (NCHW) + per-block (q-x)^2 partial sums.
// grid = 1024 blocks x 256 threads: lanes over 32 hw positions, 8 c-groups.
// ---------------------------------------------------------------------------
__global__ __launch_bounds__(256) void vq_gather_kernel(
    const float* __restrict__ inputs,
    const float* __restrict__ cb,
    const int*   __restrict__ idx,
    float* __restrict__ quantOut,
    float* __restrict__ blockSums)
{
  const int lane = threadIdx.x & 31;
  const int cgrp = threadIdx.x >> 5;         // 0..7 -> channel range cgrp*32..+31
  const int b    = blockIdx.x >> 5;          // / 32
  const int hw   = ((blockIdx.x & 31) << 5) + lane;
  const int token = b * HW + hw;
  const int e = idx[token];
  const float* cbRow = cb + (size_t)e * EMB_DIM;
  const size_t gbase = (size_t)b * EMB_DIM * HW + hw;

  float s = 0.0f;
#pragma unroll 4
  for (int i = 0; i < 32; ++i) {
    const int c = cgrp * 32 + i;
    const float q = cbRow[c];
    const float x = inputs[gbase + (size_t)c * HW];
    const float d = q - x;
    s += d * d;
    quantOut[gbase + (size_t)c * HW] = q;    // quantized_st forward value == q
  }

  __shared__ float red[256];
  red[threadIdx.x] = s;
  __syncthreads();
  for (int st = 128; st > 0; st >>= 1) {
    if (threadIdx.x < st) red[threadIdx.x] += red[threadIdx.x + st];
    __syncthreads();
  }
  if (threadIdx.x == 0) blockSums[blockIdx.x] = red[0];
}

// ---------------------------------------------------------------------------
// Kernel D: finalize loss + perplexity (deterministic fixed-order reduction).
// ---------------------------------------------------------------------------
__global__ __launch_bounds__(256) void vq_final_kernel(
    const float* __restrict__ blockSums,
    const float* __restrict__ counts,
    float* __restrict__ lossOut,
    float* __restrict__ perpOut)
{
  __shared__ float redL[256];
  __shared__ float redH[256];
  float sL = 0.0f, sH = 0.0f;
  for (int i = threadIdx.x; i < NUM_EMB; i += 256) {
    sL += blockSums[i];
    const float p = counts[i] * (1.0f / (float)N_TOKENS);
    sH += p * logf(p + 1e-10f);
  }
  redL[threadIdx.x] = sL;
  redH[threadIdx.x] = sH;
  __syncthreads();
  for (int st = 128; st > 0; st >>= 1) {
    if (threadIdx.x < st) {
      redL[threadIdx.x] += redL[threadIdx.x + st];
      redH[threadIdx.x] += redH[threadIdx.x + st];
    }
    __syncthreads();
  }
  if (threadIdx.x == 0) {
    // loss = q_latent + 0.25*e_latent; both forward values equal mean((q-x)^2)
    lossOut[0] = 1.25f * redL[0] / (float)Q_ELEMS;
    perpOut[0] = expf(-redH[0]);
  }
}

// ---------------------------------------------------------------------------
extern "C" void kernel_launch(void* const* d_in, const int* in_sizes, int n_in,
                              void* d_out, int out_size, void* d_ws, size_t ws_size,
                              hipStream_t stream) {
  (void)in_sizes; (void)n_in; (void)out_size; (void)ws_size;

  const float* inputs = (const float*)d_in[0];   // (32,256,32,32) f32
  const float* cb     = (const float*)d_in[1];   // (1024,256) f32

  float* out      = (float*)d_out;
  float* lossOut  = out;                          // [0]
  float* quantOut = out + 1;                      // [1 .. 1+Q_ELEMS)
  float* perpOut  = out + 1 + Q_ELEMS;            // [1+Q_ELEMS]
  float* idxOutF  = out + 2 + Q_ELEMS;            // [2+Q_ELEMS .. +N_TOKENS)

  char* ws = (char*)d_ws;
  uint32_t* cbP    = (uint32_t*)(ws + WS_CBP);
  float* cnorm     = (float*)(ws + WS_CNORM);
  float* counts    = (float*)(ws + WS_COUNTS);
  int*   idxI      = (int*)  (ws + WS_IDX);
  float* blockSums = (float*)(ws + WS_BSUM);

  vq_prep_kernel  <<<NUM_EMB,        128, 0, stream>>>(cb, cbP, cnorm, counts);
  vq_argmin_kernel<<<N_TOKENS / 128, 256, 0, stream>>>(inputs, cbP, cnorm,
                                                       idxI, idxOutF, counts);
  vq_gather_kernel<<<N_TOKENS / 32,  256, 0, stream>>>(inputs, cb, idxI,
                                                       quantOut, blockSums);
  vq_final_kernel <<<1,              256, 0, stream>>>(blockSums, counts,
                                                       lossOut, perpOut);
}
